// DualLearner_87488483820015
// MI455X (gfx1250) — matrix operand order repaired and checked by reading, not compiled, not run on hardware
//
#include <hip/hip_runtime.h>
#include <hip/hip_bf16.h>

// ---------------- problem constants (from reference) ----------------
#define NN   50000      // nodes  (50000 / 16 = 3125 exact row tiles)
#define EE   800000     // edges
#define INF_ 256        // in_dim
#define OUTF 128        // out_dim
#define HH   4          // heads
#define CC   32         // per-head channels

typedef __attribute__((ext_vector_type(16))) __bf16 v16bf;
typedef __attribute__((ext_vector_type(8)))  __bf16 v8bf;
typedef __attribute__((ext_vector_type(8)))  float  v8f;

// ---------------- helpers ----------------
__device__ __forceinline__ void atomicMaxFloat(float* addr, float val) {
    // order-correct float max via int punning (works with -inf init)
    if (val >= 0.f) atomicMax((int*)addr, __float_as_int(val));
    else            atomicMin((unsigned int*)addr, __float_as_uint(val));
}

// ---------------- 0) init: zero accumulators, -inf seg-max, minmax seeds ----
__global__ __launch_bounds__(256)
void init_ws(float* __restrict__ out, float* __restrict__ amax,
             float* __restrict__ denom, int* __restrict__ minmax)
{
    const int tid = blockIdx.x * 256 + threadIdx.x;
    if (tid < NN * OUTF) out[tid] = 0.f;
    if (tid < NN * HH) {
        amax[tid]  = -__builtin_huge_valf();
        denom[tid] = 0.f;
    }
    if (tid == 0) { minmax[0] = 0x7F800000; /* +inf */ minmax[1] = 0; }
}

// ---------------- 1a) pack W into bf16 WMMA-B fragment order ----------------
// Bpack[((t*8 + s)*32 + lane)*16 + j], t = 16-col tile (0..31), s = k-step.
// lane<16 : N = t*16+lane,     j -> K = s*32 + j
// lane>=16: N = t*16+lane-16,  j -> K = s*32 + 16 + j
__global__ __launch_bounds__(256)
void pack_w(const float* __restrict__ Wq, const float* __restrict__ Wk,
            const float* __restrict__ Wv, const float* __restrict__ Ws,
            unsigned short* __restrict__ Bpack)
{
    const int slot = blockIdx.x * 256 + threadIdx.x;   // 0..8191
    if (slot >= 32 * 8 * 32) return;
    const int l = slot & 31;
    const int s = (slot >> 5) & 7;
    const int t = slot >> 8;
    const int colg = t * 16 + (l & 15);
    const int wcol = colg & 127;
    const float* W;
    switch (colg >> 7) {
        case 0:  W = Wq; break;
        case 1:  W = Wk; break;
        case 2:  W = Wv; break;
        default: W = Ws; break;
    }
    const int kbase = s * 32 + (l >> 4) * 16;
    unsigned short* dst = Bpack + (size_t)slot * 16;
#pragma unroll
    for (int j = 0; j < 16; ++j)
        dst[j] = (unsigned short)(__float_as_uint(W[(kbase + j) * OUTF + wcol]) >> 16);
}

// ---------------- 1b) fused q|k|v|s projection via bf16 WMMA ----------------
// grid = (3125, 4), block = 128 (4 waves). Block: 16 rows x 128 cols of [N,512]
// (one weight matrix per block). Each wave computes two 16x16 tiles sharing
// one A fragment -> 16 static v_wmma per kernel, fully unrolled over K.
__global__ __launch_bounds__(128)
void gemm_qkvs(const float* __restrict__ x,
               const unsigned short* __restrict__ Bpack,
               const float* __restrict__ bq, const float* __restrict__ bk,
               const float* __restrict__ bv, const float* __restrict__ bs,
               float* __restrict__ qkvs)
{
    __shared__ unsigned short aSh[16][264];   // full 16x256 bf16 A tile (+pad)

    const int m0   = blockIdx.x * 16;
    const int wave = threadIdx.x >> 5;
    const int lane = threadIdx.x & 31;
    const int t0   = blockIdx.y * 8 + wave * 2;      // first of two col tiles

    // cooperative A staging: 128 threads x 32 floats, bf16-truncate, pack 2/word
    {
        const int r  = threadIdx.x >> 3;             // 0..15
        const int c0 = (threadIdx.x & 7) * 32;       // 32 consecutive K
        const float4* xp =
            reinterpret_cast<const float4*>(x + (size_t)(m0 + r) * INF_ + c0);
#pragma unroll
        for (int q = 0; q < 8; ++q) {
            const float4 f = xp[q];
            const unsigned int p0 = (__float_as_uint(f.x) >> 16) |
                                    (__float_as_uint(f.y) & 0xFFFF0000u);
            const unsigned int p1 = (__float_as_uint(f.z) >> 16) |
                                    (__float_as_uint(f.w) & 0xFFFF0000u);
            *reinterpret_cast<uint2*>(&aSh[r][c0 + q * 4]) = make_uint2(p0, p1);
        }
    }
    __syncthreads();

    const int arow = lane & 15;
    const int kb   = (lane >> 4) * 16;

    v8f acc0 = {0.f, 0.f, 0.f, 0.f, 0.f, 0.f, 0.f, 0.f};
    v8f acc1 = {0.f, 0.f, 0.f, 0.f, 0.f, 0.f, 0.f, 0.f};

#pragma unroll
    for (int s = 0; s < 8; ++s) {
        // A fragment: lane<16 K s32+0..15; lane>=16 K s32+16..31
        const v8bf lo = *reinterpret_cast<const v8bf*>(&aSh[arow][s * 32 + kb]);
        const v8bf hi = *reinterpret_cast<const v8bf*>(&aSh[arow][s * 32 + kb + 8]);
        const v16bf af = __builtin_shufflevector(lo, hi,
            0, 1, 2, 3, 4, 5, 6, 7, 8, 9, 10, 11, 12, 13, 14, 15);

        // packed B fragments: contiguous 32B per lane, coalesced
        const v16bf b0 = *reinterpret_cast<const v16bf*>(
            Bpack + ((size_t)(t0 + 0) * 8 + s) * 512 + lane * 16);
        const v16bf b1 = *reinterpret_cast<const v16bf*>(
            Bpack + ((size_t)(t0 + 1) * 8 + s) * 512 + lane * 16);

        acc0 = __builtin_amdgcn_wmma_f32_16x16x32_bf16(
                   false, af, false, b0, (short)0, acc0, false, false);
        acc1 = __builtin_amdgcn_wmma_f32_16x16x32_bf16(
                   false, af, false, b1, (short)0, acc1, false, false);
    }

    // bias select is block-uniform (128 cols == one weight matrix)
    const float* Bv;
    switch (blockIdx.y) {
        case 0:  Bv = bq; break;
        case 1:  Bv = bk; break;
        case 2:  Bv = bv; break;
        default: Bv = bs; break;
    }

    // C/D layout: VGPR g -> row g (+8 for lanes 16-31), col = lane&15
    const int rbase = m0 + (lane >> 4) * 8;
#pragma unroll
    for (int tt = 0; tt < 2; ++tt) {
        const int colg = (t0 + tt) * 16 + (lane & 15);
        const float bias = Bv[colg & 127];
        const v8f acc = tt ? acc1 : acc0;
#pragma unroll
        for (int g = 0; g < 8; ++g)
            qkvs[(size_t)(rbase + g) * 512 + colg] = acc[g] + bias;
    }
}

// ---------------- 2) per-edge attention logits + segment max ----------------
// one wave32 per edge; lane -> (head = lane>>3, float4 chunk = lane&7)
__global__ __launch_bounds__(256)
void attn_logits(const int* __restrict__ ei, const float* __restrict__ qkvs,
                 float* __restrict__ alpha, float* __restrict__ amax)
{
    const int wid  = (blockIdx.x * 256 + threadIdx.x) >> 5;   // edge id
    const int lane = threadIdx.x & 31;
    if (wid >= EE) return;
    const int src = ei[wid];
    const int dst = ei[EE + wid];
    const int h   = lane >> 3;
    const int sub = lane & 7;

    const float4 q4 = *reinterpret_cast<const float4*>(
        qkvs + (size_t)dst * 512 + 0   + h * CC + sub * 4);
    const float4 k4 = *reinterpret_cast<const float4*>(
        qkvs + (size_t)src * 512 + 128 + h * CC + sub * 4);

    float p = q4.x * k4.x + q4.y * k4.y + q4.z * k4.z + q4.w * k4.w;
    p += __shfl_xor(p, 1, 32);
    p += __shfl_xor(p, 2, 32);
    p += __shfl_xor(p, 4, 32);

    if (sub == 0) {
        const float a = p * 0.17677669529663687f;   // 1/sqrt(32)
        alpha[(size_t)wid * HH + h] = a;
        atomicMaxFloat(amax + (size_t)dst * HH + h, a);
    }
}

// ---------------- 3) exp(alpha - amax) + segment sum ------------------------
__global__ __launch_bounds__(256)
void exp_denom(const int* __restrict__ ei, float* __restrict__ alpha,
               const float* __restrict__ amax, float* __restrict__ denom)
{
    const int tid = blockIdx.x * 256 + threadIdx.x;
    if (tid >= EE * HH) return;
    const int e = tid >> 2, h = tid & 3;
    const int dst = ei[EE + e];
    const float ex = __expf(alpha[tid] - amax[(size_t)dst * HH + h]);
    alpha[tid] = ex;
    atomicAdd(denom + (size_t)dst * HH + h, ex);
}

// ---------------- 4) weighted scatter-aggregation ---------------------------
__global__ __launch_bounds__(256)
void aggregate(const int* __restrict__ ei, const float* __restrict__ qkvs,
               const float* __restrict__ alpha, const float* __restrict__ denom,
               float* __restrict__ out)
{
    const int wid  = (blockIdx.x * 256 + threadIdx.x) >> 5;
    const int lane = threadIdx.x & 31;
    if (wid >= EE) return;
    const int src = ei[wid];
    const int dst = ei[EE + wid];
    const int h   = lane >> 3;
    const int sub = lane & 7;

    const float attn = alpha[(size_t)wid * HH + h] /
                       (denom[(size_t)dst * HH + h] + 1e-16f);
    const float4 v4 = *reinterpret_cast<const float4*>(
        qkvs + (size_t)src * 512 + 256 + h * CC + sub * 4);
    float* o = out + (size_t)dst * OUTF + h * CC + sub * 4;
    atomicAdd(o + 0, attn * v4.x);
    atomicAdd(o + 1, attn * v4.y);
    atomicAdd(o + 2, attn * v4.z);
    atomicAdd(o + 3, attn * v4.w);
}

// ---------------- 5) skip connection ---------------------------------------
__global__ __launch_bounds__(256)
void skip_add(float* __restrict__ out, const float* __restrict__ qkvs)
{
    const int tid = blockIdx.x * 256 + threadIdx.x;
    if (tid >= NN * OUTF) return;
    const int n = tid >> 7, f = tid & 127;
    out[tid] += qkvs[(size_t)n * 512 + 384 + f];
}

// ---------------- 6) GraphNorm column stats (one block per column) ----------
__global__ __launch_bounds__(256)
void col_stats(const float* __restrict__ out, const float* __restrict__ mscale,
               const float* __restrict__ gnw, float* __restrict__ stats)
{
    const int c = blockIdx.x;   // 0..127
    float s1 = 0.f, s2 = 0.f;
    for (int r = threadIdx.x; r < NN; r += 256) {
        const float v = out[(size_t)r * OUTF + c];
        s1 += v; s2 += v * v;
    }
    __shared__ float sh1[256], sh2[256];
    sh1[threadIdx.x] = s1; sh2[threadIdx.x] = s2;
    __syncthreads();
    for (int off = 128; off; off >>= 1) {
        if (threadIdx.x < off) {
            sh1[threadIdx.x] += sh1[threadIdx.x + off];
            sh2[threadIdx.x] += sh2[threadIdx.x + off];
        }
        __syncthreads();
    }
    if (threadIdx.x == 0) {
        const float invN = 1.f / (float)NN;
        const float mu = sh1[0] * invN;
        const float a  = mscale[c];
        // E[(v - a*mu)^2] = S2/N - mu^2 * (2a - a^2)
        const float var = sh2[0] * invN - mu * mu * (2.f * a - a * a);
        stats[c]          = a * mu;                        // shift
        stats[OUTF + c]   = gnw[c] * rsqrtf(var + 1e-5f);  // scale
    }
}

// ---------------- 7) normalize + ReLU + global min/max ----------------------
__global__ __launch_bounds__(256)
void norm_relu_minmax(float* __restrict__ out, const float* __restrict__ stats,
                      const float* __restrict__ gnb, int* __restrict__ minmax)
{
    const int tid = blockIdx.x * 256 + threadIdx.x;
    float xmn = __builtin_huge_valf(), xmx = 0.f;
    if (tid < NN * OUTF) {
        const int c = tid & 127;
        const float y  = (out[tid] - stats[c]) * stats[OUTF + c] + gnb[c];
        const float xt = fmaxf(y, 0.f);
        out[tid] = xt;
        xmn = xt; xmx = xt;
    }
    __shared__ float smn[256], smx[256];
    smn[threadIdx.x] = xmn; smx[threadIdx.x] = xmx;
    __syncthreads();
    for (int off = 128; off; off >>= 1) {
        if (threadIdx.x < off) {
            smn[threadIdx.x] = fminf(smn[threadIdx.x], smn[threadIdx.x + off]);
            smx[threadIdx.x] = fmaxf(smx[threadIdx.x], smx[threadIdx.x + off]);
        }
        __syncthreads();
    }
    if (threadIdx.x == 0) {    // values are >= 0 -> int order == float order
        atomicMin(minmax + 0, __float_as_int(smn[0]));
        atomicMax(minmax + 1, __float_as_int(smx[0]));
    }
}

// ---------------- 8) final min-max scaling ----------------------------------
__global__ __launch_bounds__(256)
void final_scale(float* __restrict__ out, const int* __restrict__ minmax)
{
    const int tid = blockIdx.x * 256 + threadIdx.x;
    if (tid >= NN * OUTF) return;
    const float mn = __int_as_float(minmax[0]);
    const float mx = __int_as_float(minmax[1]);
    out[tid] = (out[tid] - mn) / (mx - mn + 1e-8f);
}

// ---------------- host launcher ---------------------------------------------
extern "C" void kernel_launch(void* const* d_in, const int* in_sizes, int n_in,
                              void* d_out, int out_size, void* d_ws, size_t ws_size,
                              hipStream_t stream)
{
    const float* x   = (const float*)d_in[0];
    const int*   ei  = (const int*)  d_in[1];
    const float* Wq  = (const float*)d_in[2];
    const float* bq  = (const float*)d_in[3];
    const float* Wk  = (const float*)d_in[4];
    const float* bk  = (const float*)d_in[5];
    const float* Wv  = (const float*)d_in[6];
    const float* bv  = (const float*)d_in[7];
    const float* Ws  = (const float*)d_in[8];
    const float* bs  = (const float*)d_in[9];
    const float* gnw = (const float*)d_in[10];
    const float* gnb = (const float*)d_in[11];
    const float* gms = (const float*)d_in[12];

    float* out = (float*)d_out;                      // [N, 128]

    // workspace layout: Bpack first (guarantees 32B alignment), then floats
    unsigned short* Bpack = (unsigned short*)d_ws;               // 32*8*32*16
    float* qkvs  = (float*)d_ws + 65536;                         // N*512
    float* alpha = qkvs  + (size_t)NN * 512;                     // E*4
    float* amax  = alpha + (size_t)EE * HH;                      // N*4
    float* denom = amax  + (size_t)NN * HH;                      // N*4
    float* stats = denom + (size_t)NN * HH;                      // 256
    int*   minmax = (int*)(stats + 2 * OUTF);                    // 2

    const int elemBlocks = (NN * OUTF + 255) / 256;              // 25000
    const int waveBlocks = ((EE * 32) + 255) / 256;              // 100000
    const int ehBlocks   = (EE * HH + 255) / 256;                // 12500

    init_ws<<<elemBlocks, 256, 0, stream>>>(out, amax, denom, minmax);
    pack_w <<<32,         256, 0, stream>>>(Wq, Wk, Wv, Ws, Bpack);

    dim3 ggrid(NN / 16, 4);                                      // 3125 x 4
    gemm_qkvs<<<ggrid, 128, 0, stream>>>(x, Bpack, bq, bk, bv, bs, qkvs);

    attn_logits<<<waveBlocks, 256, 0, stream>>>(ei, qkvs, alpha, amax);
    exp_denom  <<<ehBlocks,   256, 0, stream>>>(ei, alpha, amax, denom);
    aggregate  <<<waveBlocks, 256, 0, stream>>>(ei, qkvs, alpha, denom, out);
    skip_add   <<<elemBlocks, 256, 0, stream>>>(out, qkvs);
    col_stats  <<<OUTF,       256, 0, stream>>>(out, gms, gnw, stats);
    norm_relu_minmax<<<elemBlocks, 256, 0, stream>>>(out, stats, gnb, minmax);
    final_scale<<<elemBlocks, 256, 0, stream>>>(out, minmax);
}
